// CTEG_official_58523224375869
// MI455X (gfx1250) — matrix operand order repaired and checked by reading, not compiled, not run on hardware
//
#include <hip/hip_runtime.h>
#include <math.h>

// ---------------- model dims ----------------
#define BB 16
#define TT 8
#define LL 64
#define VV 32000
#define EE 300
#define HH 512
#define AA 128
#define MM 120
#define G4H 2048           // 4*H
#define KDEC 1632          // pad(1112+512) to mult of 32
#define KENC 832           // pad(300+512) to mult of 32
#define KMEM 320           // pad(300)
#define NMEM 304           // pad(300) to mult of 16
#define MROWS (BB*MM)      // 1920

typedef __attribute__((ext_vector_type(16))) __bf16 v16bf;
typedef __attribute__((ext_vector_type(8)))  float  v8f;

static __device__ __forceinline__ unsigned short f2bf(float f) {
  union { float f; unsigned u; } a; a.f = f;
  unsigned u = a.u;
  unsigned r = u + 0x7FFFu + ((u >> 16) & 1u);
  return (unsigned short)(r >> 16);
}
static __device__ __forceinline__ float sigm(float x) { return 1.0f / (1.0f + expf(-x)); }

// ---------------- generic bf16 WMMA GEMM ----------------
// Out[rtile*16 + r][ntile*16 + n] = sum_k X[row][k] * W[col][k] + bias[col]
// X: bf16 [rows][ldx], W: bf16 [N][ldw], Kp multiple of 32, one wave per 16x16 tile.
// A-frag (16x32 bf16): lane<16 -> row=l, k in {0..7}+{16..23}; lane>=16 -> row=l, k in {8..15}+{24..31}
// B-frag (32x16 bf16): lane<16 -> col=l, k in {0..7}+{16..23}; lane>=16 -> col=l, k in {8..15}+{24..31}
// C/D (16x16 f32): vgpr r -> row r + 8*(lane>=16), col = lane&15
__global__ void CTEG_wmma_gemm(const unsigned short* __restrict__ X, int ldx,
                               const unsigned short* __restrict__ W, int ldw,
                               const float* __restrict__ bias,
                               float* __restrict__ Out, long long ldo,
                               int Kp) {
  int lane = threadIdx.x & 31;
  int half = lane >> 4;
  int l = lane & 15;
  const unsigned short* xrow = X + (size_t)(blockIdx.y * 16 + l) * ldx + half * 8;
  const unsigned short* wrow = W + (size_t)(blockIdx.x * 16 + l) * ldw + half * 8;
  v8f acc = {};
  union U { uint4 q[2]; v16bf v; };
  for (int k = 0; k < Kp; k += 32) {
    U a, b;
    a.q[0] = *(const uint4*)(const void*)(xrow + k);
    a.q[1] = *(const uint4*)(const void*)(xrow + k + 16);
    b.q[0] = *(const uint4*)(const void*)(wrow + k);
    b.q[1] = *(const uint4*)(const void*)(wrow + k + 16);
    __builtin_prefetch(wrow + k + 32, 0, 0);   // global_prefetch_b8 on next W tile
    acc = __builtin_amdgcn_wmma_f32_16x16x32_bf16(false, a.v, false, b.v,
                                                  (short)0, acc, false, false);
  }
  int col = blockIdx.x * 16 + l;
  float bv = bias ? bias[col] : 0.0f;
  long long rbase = (long long)blockIdx.y * 16 + half * 8;
  for (int r = 0; r < 8; ++r)
    Out[(rbase + r) * ldo + col] = acc[r] + bv;
}

// ---------------- weight packing ----------------
__global__ void CTEG_pack_bf16(const float* __restrict__ src, int srcN, int K,
                               unsigned short* __restrict__ dst, int dstN, int ld) {
  int i = blockIdx.x * blockDim.x + threadIdx.x;
  if (i >= dstN * ld) return;
  int k = i % ld, n = i / ld;
  float v = (n < srcN && k < K) ? src[(size_t)n * K + k] : 0.0f;
  dst[i] = f2bf(v);
}

__global__ void CTEG_pack_concat(const float* __restrict__ A, int Ka,
                                 const float* __restrict__ Bm, int Kb,
                                 unsigned short* __restrict__ dst, int N, int ld) {
  int i = blockIdx.x * blockDim.x + threadIdx.x;
  if (i >= N * ld) return;
  int k = i % ld, n = i / ld;
  float v = 0.0f;
  if (k < Ka) v = A[(size_t)n * Ka + k];
  else if (k < Ka + Kb) v = Bm[(size_t)n * Kb + (k - Ka)];
  dst[i] = f2bf(v);
}

__global__ void CTEG_fill0(float* p, int n) {
  int i = blockIdx.x * blockDim.x + threadIdx.x;
  if (i < n) p[i] = 0.0f;
}

// ---------------- memory bank init ----------------
__global__ void CTEG_mem_init(const float* __restrict__ emb, const int* __restrict__ mems,
                              float* __restrict__ mem, unsigned short* __restrict__ memb) {
  int i = blockIdx.x * blockDim.x + threadIdx.x;   // MROWS*KMEM
  if (i >= MROWS * KMEM) return;
  int e = i % KMEM, row = i / KMEM;
  float v = 0.0f;
  if (e < EE) {
    int tok = mems[row];
    v = emb[(size_t)tok * EE + e];
    mem[(size_t)row * EE + e] = v;
  }
  memb[i] = f2bf(v);
}

// ---------------- encoder input pack ----------------
__global__ void CTEG_enc_pack(const float* __restrict__ emb, const int* __restrict__ topic,
                              int t, const float* __restrict__ h,
                              unsigned short* __restrict__ Xenc) {
  int i = blockIdx.x * blockDim.x + threadIdx.x;   // BB*KENC
  if (i >= BB * KENC) return;
  int b = i / KENC, k = i % KENC;
  float v;
  if (k < EE)      { int tok = topic[b * TT + t]; v = emb[(size_t)tok * EE + k]; }
  else if (k < EE + HH) v = h[b * HH + (k - EE)];
  else v = 0.0f;
  Xenc[i] = f2bf(v);
}

// ---------------- LSTM gate nonlinearity (i,f,g,o) ----------------
__global__ void CTEG_lstm_gate(const float* __restrict__ G, float* __restrict__ h,
                               float* __restrict__ c, unsigned short* hb, float* out_h) {
  int i = blockIdx.x * blockDim.x + threadIdx.x;   // BB*HH
  if (i >= BB * HH) return;
  int b = i / HH, j = i % HH;
  const float* g = G + (size_t)b * G4H;
  float ig = sigm(g[j]);
  float fg = sigm(g[HH + j]);
  float gg = tanhf(g[2 * HH + j]);
  float og = sigm(g[3 * HH + j]);
  float c2 = fg * c[i] + ig * gg;
  float h2 = og * tanhf(c2);
  c[i] = c2; h[i] = h2;
  if (hb)    hb[i] = f2bf(h2);
  if (out_h) out_h[i] = h2;
}

// ---------------- encoder combine + projection ----------------
__global__ void CTEG_enc_combine(const float* of, const float* ob,
                                 const float* hf, const float* cf,
                                 const float* hbk, const float* cbk,
                                 float* __restrict__ enc_outs,
                                 float* hdec, float* cdec, unsigned short* hdecb) {
  int i = blockIdx.x * blockDim.x + threadIdx.x;   // BB*TT*HH
  if (i >= BB * TT * HH) return;
  int hh = i % HH, t = (i / HH) % TT, b = i / (HH * TT);
  enc_outs[i] = of[(size_t)t * BB * HH + b * HH + hh] +
                ob[(size_t)(TT - 1 - t) * BB * HH + b * HH + hh];
  if (i < BB * HH) {
    float hv = hf[i] + hbk[i];
    hdec[i] = hv; cdec[i] = cf[i] + cbk[i];
    hdecb[i] = f2bf(hv);
  }
}

__global__ void CTEG_enc_proj(const float* __restrict__ enc_outs,
                              const float* __restrict__ Wep, const float* __restrict__ bep,
                              float* __restrict__ enc_proc) {
  int i = blockIdx.x * blockDim.x + threadIdx.x;   // BB*TT*AA
  if (i >= BB * TT * AA) return;
  int a = i % AA, bt = i / AA;
  const float* er = enc_outs + (size_t)bt * HH;
  const float* wr = Wep + (size_t)a * HH;
  float s = bep[a];
  for (int k = 0; k < HH; ++k) s += er[k] * wr[k];
  enc_proc[i] = s;
}

// ---------------- decoder per-step prep (attention, memory read, pack Xcat) ----------------
__global__ void CTEG_dec_prep(const float* __restrict__ emb, const int* __restrict__ essay,
                              int step, const float* __restrict__ hdec,
                              const float* __restrict__ cdec,
                              const float* __restrict__ Wp1, const float* __restrict__ bp1,
                              const float* __restrict__ Wp2, const float* __restrict__ bp2,
                              const float* __restrict__ Wi1, const float* __restrict__ bi1,
                              const float* __restrict__ bmp,
                              const float* __restrict__ mem,
                              const float* __restrict__ enc_proc,
                              const float* __restrict__ enc_outs,
                              const float* __restrict__ attn_v,
                              float* __restrict__ xproj, float* __restrict__ gate,
                              unsigned short* __restrict__ Xcat) {
  __shared__ float xs[BB][EE];
  __shared__ float vs[BB][EE];
  __shared__ float qs[BB][AA];
  __shared__ float sim[BB][MM];
  __shared__ float sc[BB][TT];
  __shared__ float aw[BB][TT];
  int tid = threadIdx.x;                    // 512 threads, 1 block

  for (int i = tid; i < BB * EE; i += 512) {
    int b = i / EE, e = i % EE;
    int tok = essay[b * LL + step];
    float x = emb[(size_t)tok * EE + e];
    xs[b][e] = x;
    Xcat[(size_t)b * KDEC + e] = f2bf(x);   // x_t part
  }
  for (int i = tid; i < BB * EE; i += 512) {        // v = tanh(h@Wp1.T + bp1)
    int b = i / EE, e = i % EE;
    float s = bp1[e];
    const float* hr = hdec + (size_t)b * HH;
    const float* wr = Wp1 + (size_t)e * HH;
    for (int k = 0; k < HH; ++k) s += hr[k] * wr[k];
    vs[b][e] = tanhf(s);
  }
  for (int i = tid; i < BB * AA; i += 512) {        // q = c@Wp2.T + bp2
    int b = i / AA, a = i % AA;
    float s = bp2[a];
    const float* cr = cdec + (size_t)b * HH;
    const float* wr = Wp2 + (size_t)a * HH;
    for (int k = 0; k < HH; ++k) s += cr[k] * wr[k];
    qs[b][a] = s;
  }
  __syncthreads();
  for (int i = tid; i < BB * EE; i += 512) {        // xproj = x@Wi1.T + bi1 + bmp
    int b = i / EE, e = i % EE;
    float s = bi1[e] + bmp[e];
    const float* wr = Wi1 + (size_t)e * EE;
    for (int k = 0; k < EE; ++k) s += xs[b][k] * wr[k];
    xproj[b * NMEM + e] = s;
  }
  for (int i = tid; i < BB * MM; i += 512) {        // gate = sigmoid(mem . x)
    int b = i / MM, m = i % MM;
    const float* mr = mem + ((size_t)b * MM + m) * EE;
    float s = 0.0f;
    for (int e = 0; e < EE; ++e) s += mr[e] * xs[b][e];
    gate[b * MM + m] = sigm(s);
  }
  for (int i = tid; i < BB * MM; i += 512) {        // sim = v . mem
    int b = i / MM, m = i % MM;
    const float* mr = mem + ((size_t)b * MM + m) * EE;
    float s = 0.0f;
    for (int e = 0; e < EE; ++e) s += mr[e] * vs[b][e];
    sim[b][m] = s;
  }
  for (int i = tid; i < BB * TT; i += 512) {        // sc = sum_a attn_v*tanh(enc_proc+q)
    int b = i / TT, t = i % TT;
    const float* ep = enc_proc + ((size_t)b * TT + t) * AA;
    float s = 0.0f;
    for (int a = 0; a < AA; ++a) s += attn_v[a] * tanhf(ep[a] + qs[b][a]);
    sc[b][t] = s;
  }
  __syncthreads();
  if (tid < BB) {                                   // softmax over memory slots
    int b = tid;
    float mx = -1e30f;
    for (int m = 0; m < MM; ++m) mx = fmaxf(mx, sim[b][m]);
    float ss = 0.0f;
    for (int m = 0; m < MM; ++m) { float e = expf(sim[b][m] - mx); sim[b][m] = e; ss += e; }
    float inv = 1.0f / ss;
    for (int m = 0; m < MM; ++m) sim[b][m] *= inv;
  } else if (tid < 2 * BB) {                        // softmax over encoder time
    int b = tid - BB;
    float mx = -1e30f;
    for (int t = 0; t < TT; ++t) mx = fmaxf(mx, sc[b][t]);
    float ss = 0.0f;
    for (int t = 0; t < TT; ++t) { float e = expf(sc[b][t] - mx); aw[b][t] = e; ss += e; }
    float inv = 1.0f / ss;
    for (int t = 0; t < TT; ++t) aw[b][t] *= inv;
  }
  __syncthreads();
  for (int i = tid; i < BB * EE; i += 512) {        // mt -> Xcat[812..1111]
    int b = i / EE, e = i % EE;
    float s = 0.0f;
    for (int m = 0; m < MM; ++m) s += sim[b][m] * mem[((size_t)b * MM + m) * EE + e];
    Xcat[(size_t)b * KDEC + EE + HH + e] = f2bf(s);
  }
  for (int i = tid; i < BB * HH; i += 512) {        // attn_val -> Xcat[300..811]
    int b = i / HH, hh = i % HH;
    float s = 0.0f;
    for (int t = 0; t < TT; ++t) s += aw[b][t] * enc_outs[((size_t)b * TT + t) * HH + hh];
    Xcat[(size_t)b * KDEC + EE + hh] = f2bf(s);
  }
  for (int i = tid; i < BB * (HH + 8); i += 512) {  // h tail + zero pad
    int b = i / (HH + 8), k = i % (HH + 8);
    float v = (k < HH) ? hdec[b * HH + k] : 0.0f;
    Xcat[(size_t)b * KDEC + 1112 + k] = f2bf(v);
  }
}

// ---------------- memory write (gated update) ----------------
__global__ void CTEG_mem_update(const float* __restrict__ candG, const float* __restrict__ xproj,
                                const float* __restrict__ gate,
                                float* __restrict__ mem, unsigned short* __restrict__ memb) {
  int i = blockIdx.x * blockDim.x + threadIdx.x;   // MROWS*EE
  if (i >= MROWS * EE) return;
  int e = i % EE, row = i / EE, b = row / MM;
  float g = gate[row];
  float cand = candG[(size_t)row * NMEM + e] + xproj[b * NMEM + e];
  float m = mem[i];
  float nm = (1.0f - g) * m + g * cand;
  if (!(nm - nm == 0.0f)) nm = 1e-40f;             // !isfinite -> EPS
  mem[i] = nm;
  memb[(size_t)row * KMEM + e] = f2bf(nm);
}

// ---------------- launch ----------------
extern "C" void kernel_launch(void* const* d_in, const int* in_sizes, int n_in,
                              void* d_out, int out_size, void* d_ws, size_t ws_size,
                              hipStream_t stream) {
  const int*   topic    = (const int*)d_in[0];
  const int*   essay    = (const int*)d_in[2];
  const int*   mems     = (const int*)d_in[4];
  const float* emb      = (const float*)d_in[5];
  const float* eWihF    = (const float*)d_in[6];
  const float* eWhhF    = (const float*)d_in[7];
  const float* ebF      = (const float*)d_in[8];
  const float* eWihB    = (const float*)d_in[9];
  const float* eWhhB    = (const float*)d_in[10];
  const float* ebB      = (const float*)d_in[11];
  const float* dWih     = (const float*)d_in[12];
  const float* dWhh     = (const float*)d_in[13];
  const float* db       = (const float*)d_in[14];
  const float* Wp1      = (const float*)d_in[15];
  const float* bp1      = (const float*)d_in[16];
  const float* Wp2      = (const float*)d_in[17];
  const float* bp2      = (const float*)d_in[18];
  const float* Wi1      = (const float*)d_in[19];
  const float* bi1      = (const float*)d_in[20];
  const float* Wmp      = (const float*)d_in[21];
  const float* bmp      = (const float*)d_in[22];
  const float* Wep      = (const float*)d_in[23];
  const float* bep      = (const float*)d_in[24];
  const float* attnv    = (const float*)d_in[25];
  const float* Wout     = (const float*)d_in[26];
  const float* bout     = (const float*)d_in[27];
  float* out = (float*)d_out;

  // workspace layout (256B aligned)
  char* base = (char*)d_ws;
  size_t off = 0;
  auto alloc = [&](size_t bytes) { size_t o = off; off = (off + bytes + 255) & ~(size_t)255; return o; };
  unsigned short* WoutB  = (unsigned short*)(base + alloc((size_t)VV * HH * 2));
  unsigned short* WdecB  = (unsigned short*)(base + alloc((size_t)G4H * KDEC * 2));
  unsigned short* WencFB = (unsigned short*)(base + alloc((size_t)G4H * KENC * 2));
  unsigned short* WencBB = (unsigned short*)(base + alloc((size_t)G4H * KENC * 2));
  unsigned short* WmpB   = (unsigned short*)(base + alloc((size_t)NMEM * KMEM * 2));
  unsigned short* Xenc   = (unsigned short*)(base + alloc((size_t)BB * KENC * 2));
  float* Genc   = (float*)(base + alloc((size_t)BB * G4H * 4));
  float* hencF  = (float*)(base + alloc((size_t)BB * HH * 4));
  float* cencF  = (float*)(base + alloc((size_t)BB * HH * 4));
  float* hencB  = (float*)(base + alloc((size_t)BB * HH * 4));
  float* cencB  = (float*)(base + alloc((size_t)BB * HH * 4));
  float* outsF  = (float*)(base + alloc((size_t)TT * BB * HH * 4));
  float* outsB  = (float*)(base + alloc((size_t)TT * BB * HH * 4));
  float* encO   = (float*)(base + alloc((size_t)BB * TT * HH * 4));
  float* encP   = (float*)(base + alloc((size_t)BB * TT * AA * 4));
  float* hdec   = (float*)(base + alloc((size_t)BB * HH * 4));
  float* cdec   = (float*)(base + alloc((size_t)BB * HH * 4));
  unsigned short* hdecB = (unsigned short*)(base + alloc((size_t)BB * HH * 2));
  float* mem    = (float*)(base + alloc((size_t)MROWS * EE * 4));
  unsigned short* memB  = (unsigned short*)(base + alloc((size_t)MROWS * KMEM * 2));
  float* candG  = (float*)(base + alloc((size_t)MROWS * NMEM * 4));
  float* xproj  = (float*)(base + alloc((size_t)BB * NMEM * 4));
  float* gate   = (float*)(base + alloc((size_t)BB * MM * 4));
  unsigned short* Xcat = (unsigned short*)(base + alloc((size_t)BB * KDEC * 2));
  float* Gdec   = (float*)(base + alloc((size_t)BB * G4H * 4));
  (void)ws_size; (void)n_in; (void)in_sizes; (void)out_size;

  // ---- weight packing (bf16) ----
  CTEG_pack_bf16<<<(VV * HH + 255) / 256, 256, 0, stream>>>(Wout, VV, HH, WoutB, VV, HH);
  CTEG_pack_concat<<<(G4H * KDEC + 255) / 256, 256, 0, stream>>>(dWih, 1112, dWhh, HH, WdecB, G4H, KDEC);
  CTEG_pack_concat<<<(G4H * KENC + 255) / 256, 256, 0, stream>>>(eWihF, EE, eWhhF, HH, WencFB, G4H, KENC);
  CTEG_pack_concat<<<(G4H * KENC + 255) / 256, 256, 0, stream>>>(eWihB, EE, eWhhB, HH, WencBB, G4H, KENC);
  CTEG_pack_bf16<<<(NMEM * KMEM + 255) / 256, 256, 0, stream>>>(Wmp, EE, EE, WmpB, NMEM, KMEM);
  CTEG_mem_init<<<(MROWS * KMEM + 255) / 256, 256, 0, stream>>>(emb, mems, mem, memB);

  // ---- encoder (bidirectional LSTM, WMMA gates) ----
  for (int dir = 0; dir < 2; ++dir) {
    float* h = dir ? hencB : hencF;
    float* c = dir ? cencB : cencF;
    float* outs = dir ? outsB : outsF;
    const unsigned short* Wb = dir ? WencBB : WencFB;
    const float* bb = dir ? ebB : ebF;
    CTEG_fill0<<<(BB * HH + 255) / 256, 256, 0, stream>>>(h, BB * HH);
    CTEG_fill0<<<(BB * HH + 255) / 256, 256, 0, stream>>>(c, BB * HH);
    for (int t = 0; t < TT; ++t) {
      int tact = dir ? (TT - 1 - t) : t;
      CTEG_enc_pack<<<(BB * KENC + 255) / 256, 256, 0, stream>>>(emb, topic, tact, h, Xenc);
      CTEG_wmma_gemm<<<dim3(G4H / 16, 1), 32, 0, stream>>>(Xenc, KENC, Wb, KENC, bb,
                                                           Genc, (long long)G4H, KENC);
      CTEG_lstm_gate<<<(BB * HH + 255) / 256, 256, 0, stream>>>(Genc, h, c, (unsigned short*)nullptr,
                                                                outs + (size_t)t * BB * HH);
    }
  }
  CTEG_enc_combine<<<(BB * TT * HH + 255) / 256, 256, 0, stream>>>(outsF, outsB, hencF, cencF,
                                                                   hencB, cencB, encO, hdec, cdec, hdecB);
  CTEG_enc_proj<<<(BB * TT * AA + 255) / 256, 256, 0, stream>>>(encO, Wep, bep, encP);

  // ---- decoder: 64 steps ----
  for (int step = 0; step < LL; ++step) {
    CTEG_dec_prep<<<1, 512, 0, stream>>>(emb, essay, step, hdec, cdec, Wp1, bp1, Wp2, bp2,
                                         Wi1, bi1, bmp, mem, encP, encO, attnv,
                                         xproj, gate, Xcat);
    // cand partial: mem @ Wmp^T  (1920x300x300 WMMA GEMM)
    CTEG_wmma_gemm<<<dim3(NMEM / 16, MROWS / 16), 32, 0, stream>>>(memB, KMEM, WmpB, KMEM,
                                                                   (const float*)nullptr,
                                                                   candG, (long long)NMEM, KMEM);
    CTEG_mem_update<<<(MROWS * EE + 255) / 256, 256, 0, stream>>>(candG, xproj, gate, mem, memB);
    // decoder LSTM gates: Xcat[16x1632] @ [Wih|Whh]^T + b
    CTEG_wmma_gemm<<<dim3(G4H / 16, 1), 32, 0, stream>>>(Xcat, KDEC, WdecB, KDEC, db,
                                                         Gdec, (long long)G4H, KDEC);
    CTEG_lstm_gate<<<(BB * HH + 255) / 256, 256, 0, stream>>>(Gdec, hdec, cdec, hdecB, (float*)nullptr);
    // logits: h2[16x512] @ Wout^T + bout -> d_out[b][step][:]
    CTEG_wmma_gemm<<<dim3(VV / 16, 1), 32, 0, stream>>>(hdecB, HH, WoutB, HH, bout,
                                                        out + (size_t)step * VV,
                                                        (long long)LL * VV, HH);
  }
}
